// MAPLE_19318762897545
// MI455X (gfx1250) — matrix operand order repaired and checked by reading, not compiled, not run on hardware
//
#include <hip/hip_runtime.h>
#include <stdint.h>

// Problem constants (match reference)
#define B_    4096
#define T_    64
#define D_    76
#define DP_   96      // D padded to multiple of 32 for bf16 WMMA K-dim
#define H_    128
#define G3_   384     // 3*H
#define K_    12
#define ITERS_ 25
#define BT_   (B_*T_)

typedef __attribute__((ext_vector_type(16))) __bf16        v16bf;
typedef __attribute__((ext_vector_type(8)))  float         v8f;
typedef __attribute__((ext_vector_type(4)))  unsigned int  u32x4;

union FragBF { v16bf v; u32x4 u[2]; };

__device__ __forceinline__ unsigned short f2bf(float x) {
  unsigned int u = __float_as_uint(x);
  u = (u + 0x7FFFu + ((u >> 16) & 1u)) >> 16;   // round-to-nearest-even
  return (unsigned short)u;
}
__device__ __forceinline__ float bf2f(unsigned short h) {
  return __uint_as_float(((unsigned int)h) << 16);
}
__device__ __forceinline__ float sigmoidf_(float x) {
  return 1.0f / (1.0f + __expf(-x));
}
__device__ __forceinline__ float urand01(uint64_t s) {  // splitmix64 -> (0,1)
  s += 0x9E3779B97F4A7C15ull;
  s = (s ^ (s >> 30)) * 0xBF58476D1CE4E5B9ull;
  s = (s ^ (s >> 27)) * 0x94D049BB133111EBull;
  s ^= s >> 31;
  return ((float)(s >> 40) + 0.5f) * (1.0f / 16777216.0f);
}

// ---------------------------------------------------------------------------
// WMMA GEMM: C[M,N](f32 or bf16) = A[M,Kd](bf16,row-major) * B[Kd,N](bf16) + bias[N]
// 128 threads = 4 waves per block; each wave owns a 16x32 C tile (two adjacent
// 16x16 N-subtiles sharing one A fragment -> 2 WMMAs per A load).
// A fragment (16x32 bf16): lane&15 = M row; lane>>4 selects K halves
//   {0..7,16..23} vs {8..15,24..31}; two 16B loads per lane.
// B fragment (32x16 bf16): lane&15 = N col; lane>>4 picks the same K halves as A (16 row-strided loads).
// C/D: 8 VGPRs, lanes 0-15 rows M+0..7 (N=lane&15), lanes 16-31 rows M+8..15.
// ---------------------------------------------------------------------------
template <bool OUT_BF16>
__global__ void wmma_gemm_bf16_kernel(const unsigned short* __restrict__ A,
                                      const unsigned short* __restrict__ Bm,
                                      const float* __restrict__ bias,
                                      void* __restrict__ Cout,
                                      int M, int N, int Kd, int lda, int ldb) {
  const int lane  = threadIdx.x & 31;
  const int wave  = threadIdx.x >> 5;
  const int tileN = blockIdx.x << 5;                 // 32-wide N tile
  const int tileM = ((blockIdx.y << 2) + wave) << 4; // 16-high M tile
  if (tileM >= M) return;                 // wave-uniform; EXEC stays all-ones

  const int row  = tileM + (lane & 15);
  const int koff = (lane >> 4) << 3;      // 0 or 8
  v8f acc0 = {}, acc1 = {};
  const unsigned short* ap = A  + (size_t)row  * lda + koff;
  const unsigned short* bp = Bm + (size_t)koff * ldb + tileN + (lane & 15);
  for (int k0 = 0; k0 < Kd; k0 += 32) {
    FragBF fa, fb0, fb1;
    fa.u[0]  = *(const u32x4*)(ap);        // K = k0+koff    .. +7
    fa.u[1]  = *(const u32x4*)(ap + 16);   // K = k0+16+koff .. +7
#pragma unroll
    for (int s = 0; s < 8; ++s) {
      fb0.v[s]     = __builtin_bit_cast(__bf16, bp[(size_t)s * ldb]);
      fb0.v[s + 8] = __builtin_bit_cast(__bf16, bp[(size_t)(16 + s) * ldb]);
      fb1.v[s]     = __builtin_bit_cast(__bf16, bp[(size_t)s * ldb + 16]);
      fb1.v[s + 8] = __builtin_bit_cast(__bf16, bp[(size_t)(16 + s) * ldb + 16]);
    }
    ap += 32;
    bp += (size_t)32 * ldb;
    __builtin_prefetch(ap, 0, 1);          // global_prefetch_b8, next K slab
    acc0 = __builtin_amdgcn_wmma_f32_16x16x32_bf16(
        false, fa.v, false, fb0.v, (short)0, acc0, false, false);
    acc1 = __builtin_amdgcn_wmma_f32_16x16x32_bf16(
        false, fa.v, false, fb1.v, (short)0, acc1, false, false);
  }
  const int   cn  = lane & 15;
  const int   cm  = (lane >> 4) << 3;
  const float bv0 = bias[tileN + cn];
  const float bv1 = bias[tileN + 16 + cn];
#pragma unroll
  for (int r = 0; r < 8; ++r) {
    float  v0  = acc0[r] + bv0;
    float  v1  = acc1[r] + bv1;
    size_t off = (size_t)(tileM + cm + r) * N + tileN + cn;
    if (OUT_BF16) {
      ((unsigned short*)Cout)[off]      = f2bf(v0);
      ((unsigned short*)Cout)[off + 16] = f2bf(v1);
    } else {
      ((float*)Cout)[off]      = v0;
      ((float*)Cout)[off + 16] = v1;
    }
  }
}

// ------------------------- small prep kernels ------------------------------
__global__ void pack_x_kernel(const float* __restrict__ x,
                              unsigned short* __restrict__ xbf) {
  size_t e = (size_t)blockIdx.x * 256 + threadIdx.x;
  if (e >= (size_t)BT_ * DP_) return;
  int c = (int)(e % DP_);
  size_t r = e / DP_;
  xbf[e] = (c < D_) ? f2bf(x[r * D_ + c]) : (unsigned short)0;
}

// Wc = proj_w @ w_ih  (folded input weights), bf16, rows D_..DP_-1 zero
__global__ void fold_w_kernel(const float* __restrict__ proj_w,
                              const float* __restrict__ w_ih,
                              unsigned short* __restrict__ Wc) {
  int e = blockIdx.x * 256 + threadIdx.x;
  if (e >= DP_ * G3_) return;
  int i = e / G3_, j = e % G3_;
  float s = 0.f;
  if (i < D_)
    for (int h = 0; h < H_; ++h) s += proj_w[i * H_ + h] * w_ih[h * G3_ + j];
  Wc[e] = (i < D_) ? f2bf(s) : (unsigned short)0;
}

// bc = proj_b @ w_ih + b_ih
__global__ void fold_b_kernel(const float* __restrict__ proj_b,
                              const float* __restrict__ w_ih,
                              const float* __restrict__ b_ih,
                              float* __restrict__ bc) {
  int j = threadIdx.x;
  if (j >= G3_) return;
  float s = b_ih[j];
  for (int h = 0; h < H_; ++h) s += proj_b[h] * w_ih[h * G3_ + j];
  bc[j] = s;
}

__global__ void conv_bf16_kernel(const float* __restrict__ src,
                                 unsigned short* __restrict__ dst, int n) {
  int e = blockIdx.x * 256 + threadIdx.x;
  if (e < n) dst[e] = f2bf(src[e]);
}

// ------------------------- GRU elementwise step ----------------------------
// GI is bf16 [BT,384] with row index b*T + t (matches x layout [B,T,D]).
__global__ void gru_update_kernel(const unsigned short* __restrict__ GI,
                                  const float* __restrict__ GH,
                                  const float* __restrict__ hprev,
                                  float* __restrict__ hcur,
                                  unsigned short* __restrict__ hbf,
                                  int t) {
  int i = blockIdx.x * 256 + threadIdx.x;     // 0 .. B*H-1
  int b = i >> 7, j = i & 127;
  size_t gib = ((size_t)b * T_ + t) * G3_;
  size_t ghb = (size_t)b * G3_;
  float ir  = bf2f(GI[gib + j]);
  float iz  = bf2f(GI[gib + H_ + j]);
  float in_ = bf2f(GI[gib + 2 * H_ + j]);
  float hr = GH[ghb + j];
  float hz = GH[ghb + H_ + j];
  float hn = GH[ghb + 2 * H_ + j];
  float ho = hprev ? hprev[i] : 0.f;
  float r = sigmoidf_(ir + hr);
  float z = sigmoidf_(iz + hz);
  float n = tanhf(in_ + r * hn);
  float hnew = (1.f - z) * n + z * ho;
  hcur[i] = hnew;
  hbf[i]  = f2bf(hnew);
}

// ------------------------- k-means (all T slices per launch) ---------------
__global__ void centers_init_kernel(const float* __restrict__ hs,
                                    float* __restrict__ centers) {
  int t = blockIdx.x, tid = threadIdx.x;
  for (int e = tid; e < K_ * H_; e += 256) {
    int k = e / H_, j = e % H_;
    uint64_t s = ((uint64_t)(t + 1) << 32) ^ (uint64_t)(k * 0x85EBCA6Bu);
    s += 0x9E3779B97F4A7C15ull;
    s = (s ^ (s >> 30)) * 0xBF58476D1CE4E5B9ull;
    s = (s ^ (s >> 27)) * 0x94D049BB133111EBull;
    s ^= s >> 31;
    int idx = (int)(s & (uint64_t)(B_ - 1));   // B_ is power of two
    centers[(size_t)t * K_ * H_ + e] = hs[((size_t)t * B_ + idx) * H_ + j];
  }
}

// One thread per row: argmin distance via ||c||^2 - 2 h.c ; accumulate sums
// per cluster into LDS, flush once per block with global atomics.
__global__ void kmeans_assign_accum_kernel(const float* __restrict__ hs,
                                           const float* __restrict__ centers,
                                           float* __restrict__ accum,
                                           float* __restrict__ cnt) {
  const int t = blockIdx.y;
  const int tid = threadIdx.x;
  const int b = blockIdx.x * 256 + tid;
  __shared__ float sc[K_ * H_];
  __shared__ float sa[K_ * H_];
  __shared__ float sn2[K_];
  __shared__ float scnt[K_];
  const float* ct = centers + (size_t)t * K_ * H_;
  for (int e = tid; e < K_ * H_; e += 256) { sc[e] = ct[e]; sa[e] = 0.f; }
  if (tid < K_) scnt[tid] = 0.f;
  __syncthreads();
  if (tid < K_) {
    float s = 0.f;
    for (int j = 0; j < H_; ++j) { float c = sc[tid * H_ + j]; s += c * c; }
    sn2[tid] = s;
  }
  __syncthreads();

  const float* h = hs + ((size_t)t * B_ + b) * H_;
  float dot[K_];
#pragma unroll
  for (int k = 0; k < K_; ++k) dot[k] = 0.f;
  for (int j = 0; j < H_; ++j) {
    float hv = h[j];
#pragma unroll
    for (int k = 0; k < K_; ++k) dot[k] += hv * sc[k * H_ + j];
  }
  int kb = 0; float best = 3.0e38f;
#pragma unroll
  for (int k = 0; k < K_; ++k) {
    float d = sn2[k] - 2.f * dot[k];
    if (d < best) { best = d; kb = k; }
  }
  for (int j = 0; j < H_; ++j) atomicAdd(&sa[kb * H_ + j], h[j]);
  atomicAdd(&scnt[kb], 1.f);
  __syncthreads();

  float* ga = accum + (size_t)t * K_ * H_;
  for (int e = tid; e < K_ * H_; e += 256) atomicAdd(&ga[e], sa[e]);
  if (tid < K_) atomicAdd(&cnt[t * K_ + tid], scnt[tid]);
}

// centers = accum / max(cnt, 1)  (empty clusters keep sum/1); re-zero accum.
__global__ void kmeans_finalize_kernel(float* __restrict__ centers,
                                       float* __restrict__ accum,
                                       float* __restrict__ cnt) {
  int t = blockIdx.x, tid = threadIdx.x;
  float* ct = centers + (size_t)t * K_ * H_;
  float* ga = accum + (size_t)t * K_ * H_;
  for (int e = tid; e < K_ * H_; e += 256) {
    int k = e / H_;
    float c = cnt[t * K_ + k];
    float denom = (c > 0.5f) ? c : 1.0f;
    ct[e] = ga[e] / denom;
    ga[e] = 0.f;
  }
  __syncthreads();
  if (tid < K_) cnt[t * K_ + tid] = 0.f;
}

// ------------------------- tiny GCN (adj = I at epoch 0) -------------------
__global__ void gcn_kernel(const float* __restrict__ centers,
                           const float* __restrict__ g1w, const float* __restrict__ g1b,
                           const float* __restrict__ g2w, const float* __restrict__ g2b,
                           float* __restrict__ h2g) {
  int t = blockIdx.x, tid = threadIdx.x;
  __shared__ float s1[K_ * H_];
  const float* ct = centers + (size_t)t * K_ * H_;
  for (int e = tid; e < K_ * H_; e += 256) {
    int k = e / H_, j = e % H_;
    float s = g1b[j];
    for (int i = 0; i < H_; ++i) s += ct[k * H_ + i] * g1w[i * H_ + j];
    s1[e] = fmaxf(s, 0.f);
  }
  __syncthreads();
  for (int e = tid; e < K_ * H_; e += 256) {
    int k = e / H_, j = e % H_;
    float s = g2b[j];
    for (int i = 0; i < H_; ++i) s += s1[k * H_ + i] * g2w[i * H_ + j];
    h2g[(size_t)t * K_ * H_ + e] = fmaxf(s, 0.f);
  }
}

// ------------------------- GRASP output: wave per (t,b) row ----------------
__global__ void grasp_kernel(const float* __restrict__ hs,
                             const float* __restrict__ centers,
                             const float* __restrict__ h2g,
                             const float* __restrict__ w1w, const float* __restrict__ w1b,
                             const float* __restrict__ w2w, const float* __restrict__ w2b,
                             float* __restrict__ out) {
  const int t = blockIdx.y;
  const int lane = threadIdx.x & 31;
  const int wave = threadIdx.x >> 5;
  const int b = (blockIdx.x << 3) + wave;

  const float* h = hs + ((size_t)t * B_ + b) * H_;
  float hv[4];
#pragma unroll
  for (int q = 0; q < 4; ++q) hv[q] = h[lane + q * 32];

  const float* ct = centers + (size_t)t * K_ * H_;
  float ek[K_];
#pragma unroll
  for (int k = 0; k < K_; ++k) {
    float p = 0.f;
#pragma unroll
    for (int q = 0; q < 4; ++q) p += hv[q] * ct[k * H_ + lane + q * 32];
    for (int off = 16; off > 0; off >>= 1) p += __shfl_xor(p, off, 32);
    ek[k] = fmaxf(p, 0.f);                       // e = relu(h @ centers.T)
  }
  // gumbel-softmax hard forward == one-hot of argmax(e + gumbel)
  int kb = 0; float besty = -3.0e38f;
#pragma unroll
  for (int k = 0; k < K_; ++k) {
    uint64_t seed = 0x2A00000000ull ^ (((uint64_t)(t * B_ + b)) << 8) ^ (uint64_t)k;
    float u = urand01(seed);
    float g = -__logf(-__logf(u + 1e-20f) + 1e-20f);
    float y = ek[k] + g;
    if (y > besty) { besty = y; kb = k; }        // uniform across wave
  }
  const float* h2 = h2g + ((size_t)t * K_ + kb) * H_;
  float cl[4];
#pragma unroll
  for (int q = 0; q < 4; ++q) cl[q] = h2[lane + q * 32];

  float p1 = 0.f, p2 = 0.f;
#pragma unroll
  for (int q = 0; q < 4; ++q) {
    p1 += cl[q] * w1w[lane + q * 32];
    p2 += hv[q] * w2w[lane + q * 32];
  }
  for (int off = 16; off > 0; off >>= 1) {
    p1 += __shfl_xor(p1, off, 32);
    p2 += __shfl_xor(p2, off, 32);
  }
  float w1v = sigmoidf_(p1 + w1b[0]);
  float w2v = sigmoidf_(p2 + w2b[0]);
  float w1n = w1v / (w1v + w2v);
#pragma unroll
  for (int q = 0; q < 4; ++q)
    out[((size_t)b * T_ + t) * H_ + lane + q * 32] =
        w1n * cl[q] + (1.f - w1n) * hv[q];
}

// ---------------------------------------------------------------------------
extern "C" void kernel_launch(void* const* d_in, const int* in_sizes, int n_in,
                              void* d_out, int out_size, void* d_ws, size_t ws_size,
                              hipStream_t stream) {
  (void)in_sizes; (void)n_in; (void)out_size; (void)ws_size;
  const float* x      = (const float*)d_in[0];
  /* d_in[1] = epoch (0 -> adj = I branch) */
  const float* proj_w = (const float*)d_in[2];
  const float* proj_b = (const float*)d_in[3];
  const float* w_ih   = (const float*)d_in[4];
  const float* w_hh   = (const float*)d_in[5];
  const float* b_ih   = (const float*)d_in[6];
  const float* b_hh   = (const float*)d_in[7];
  const float* g1w    = (const float*)d_in[8];
  const float* g1b    = (const float*)d_in[9];
  const float* g2w    = (const float*)d_in[10];
  const float* g2b    = (const float*)d_in[11];
  const float* w1w    = (const float*)d_in[12];
  const float* w1b    = (const float*)d_in[13];
  const float* w2w    = (const float*)d_in[14];
  const float* w2b    = (const float*)d_in[15];
  float* out = (float*)d_out;

  char* p = (char*)d_ws;
  auto carve = [&](size_t bytes) {
    char* r = p; p += (bytes + 255) & ~(size_t)255; return r;
  };
  unsigned short* xbf = (unsigned short*)carve((size_t)BT_ * DP_ * 2);
  unsigned short* Wc  = (unsigned short*)carve((size_t)DP_ * G3_ * 2);
  float*          bc  = (float*)carve((size_t)G3_ * 4);
  unsigned short* whh = (unsigned short*)carve((size_t)H_ * G3_ * 2);
  unsigned short* GI  = (unsigned short*)carve((size_t)BT_ * G3_ * 2);
  float*          hs  = (float*)carve((size_t)T_ * B_ * H_ * 4);
  float*          GH  = (float*)carve((size_t)B_ * G3_ * 4);
  unsigned short* hbf = (unsigned short*)carve((size_t)B_ * H_ * 2);
  float*      centers = (float*)carve((size_t)T_ * K_ * H_ * 4);
  float*        accum = (float*)carve((size_t)T_ * K_ * H_ * 4);
  float*          cnt = (float*)carve((size_t)T_ * K_ * 4);
  float*          h2g = (float*)carve((size_t)T_ * K_ * H_ * 4);

  hipMemsetAsync(hbf,   0, (size_t)B_ * H_ * 2,      stream);  // h0 = 0
  hipMemsetAsync(accum, 0, (size_t)T_ * K_ * H_ * 4, stream);
  hipMemsetAsync(cnt,   0, (size_t)T_ * K_ * 4,      stream);

  // Prep: pad/convert x; fold proj into the input-gate weights (gi = x@Wc + bc)
  pack_x_kernel<<<(BT_ * DP_ + 255) / 256, 256, 0, stream>>>(x, xbf);
  fold_w_kernel<<<(DP_ * G3_ + 255) / 256, 256, 0, stream>>>(proj_w, w_ih, Wc);
  fold_b_kernel<<<1, G3_, 0, stream>>>(proj_b, w_ih, b_ih, bc);
  conv_bf16_kernel<<<(H_ * G3_ + 255) / 256, 256, 0, stream>>>(w_hh, whh, H_ * G3_);

  // Big upfront WMMA GEMM: GI[BT,384] = x_pad @ Wc + bc   (bf16 out)
  wmma_gemm_bf16_kernel<true><<<dim3(G3_ / 32, BT_ / 64), 128, 0, stream>>>(
      xbf, Wc, bc, GI, BT_, G3_, DP_, DP_, G3_);

  // Sequential GRU: per step one WMMA GEMM (h @ w_hh + b_hh) + gate kernel
  for (int t = 0; t < T_; ++t) {
    wmma_gemm_bf16_kernel<false><<<dim3(G3_ / 32, B_ / 64), 128, 0, stream>>>(
        hbf, whh, b_hh, GH, B_, G3_, H_, H_, G3_);
    gru_update_kernel<<<(B_ * H_) / 256, 256, 0, stream>>>(
        GI, GH, (t == 0) ? (const float*)nullptr : hs + (size_t)(t - 1) * B_ * H_,
        hs + (size_t)t * B_ * H_, hbf, t);
  }

  // k-means over all 64 timestep slices at once
  centers_init_kernel<<<T_, 256, 0, stream>>>(hs, centers);
  for (int it = 0; it < ITERS_; ++it) {
    kmeans_assign_accum_kernel<<<dim3(B_ / 256, T_), 256, 0, stream>>>(
        hs, centers, accum, cnt);
    kmeans_finalize_kernel<<<T_, 256, 0, stream>>>(centers, accum, cnt);
  }

  gcn_kernel<<<T_, 256, 0, stream>>>(centers, g1w, g1b, g2w, g2b, h2g);
  grasp_kernel<<<dim3(B_ / 8, T_), 256, 0, stream>>>(
      hs, centers, h2g, w1w, w1b, w2w, w2b, out);
}
